// CharEmbedder_23304492548444
// MI455X (gfx1250) — matrix operand order repaired and checked by reading, not compiled, not run on hardware
//
#include <hip/hip_runtime.h>
#include <hip/hip_bf16.h>

// ---------------------------------------------------------------------------
// ELMo char-CNN encoder for MI455X (gfx1250, wave32, WMMA bf16).
// Conv (implicit GEMM), highway x2, projection all run on
// v_wmma_f32_16x16x32_bf16 with f32 accumulation. Kernels are template-
// specialized so fragment arrays are statically indexed (no v_movrels) and
// K-loops unroll with independent accumulator chains.
// ---------------------------------------------------------------------------

typedef __attribute__((ext_vector_type(16))) __bf16 v16bf;
typedef __attribute__((ext_vector_type(8)))  __bf16 v8bf;
typedef __attribute__((ext_vector_type(8)))  float  v8f;

#define TOKENS   2048          // BATCH * SEQ
#define NCH      50            // MAX_CHARS
#define EMB      16
#define XCOL     832           // 50*16 = 800, padded to 832 (zeros)
#define NFILT    2048
#define HWN      4096          // 2 * NFILT
#define ODIM     512
#define CONVW_ELEMS 219136     // total packed conv-weight elements

// ---------------- WMMA fragment helpers (contiguous K order, k = 16*half + s) ----------------

__device__ __forceinline__ v16bf make16(v8bf lo, v8bf hi) {
    return __builtin_shufflevector(lo, hi, 0,1,2,3,4,5,6,7,8,9,10,11,12,13,14,15);
}

// A fragment 16x32 bf16: lane = M row; elements 0..15 = K[colBase + 16*half ..]
// (16 contiguous K). Two 16B loads.
__device__ __forceinline__ v16bf load_a_frag(const __bf16* __restrict__ row,
                                             int colBase, int half) {
    v8bf lo = *(const v8bf*)(row + colBase + half * 16);
    v8bf hi = *(const v8bf*)(row + colBase + half * 16 + 8);
    return make16(lo, hi);
}

// B fragment 32x16 bf16 from Bt stored [N][K] (K contiguous per filter):
// lane = N column (n passed in already includes lane&15), 16 contiguous K
// starting at k0 + 16*half. Two 16B loads.
__device__ __forceinline__ v16bf load_b_frag(const __bf16* __restrict__ bt,
                                             int ldb, int n, int k0, int half) {
    const __bf16* p = bt + (size_t)n * ldb + k0 + half * 16;
    v8bf lo = *(const v8bf*)(p);
    v8bf hi = *(const v8bf*)(p + 8);
    return make16(lo, hi);
}

__device__ __forceinline__ v8f wmma_bf16(v16bf a, v16bf b, v8f c) {
    // (neg_a, A, neg_b, B, c_mod, C, reuse_a, reuse_b)
    return __builtin_amdgcn_wmma_f32_16x16x32_bf16(false, a, false, b,
                                                   (short)0, c, false, false);
}

// ---------------- Prep kernels (bandwidth-trivial) ---------------------------

// Embedding lookup -> Xb[t][c*16+e] bf16, zero-padded cols [800,832).
__global__ void k_prep_x(const int* __restrict__ ids,
                         const float* __restrict__ emb,
                         __bf16* __restrict__ Xb) {
    int idx = blockIdx.x * blockDim.x + threadIdx.x;
    if (idx >= TOKENS * XCOL) return;
    int t = idx / XCOL, col = idx - t * XCOL;
    float v = 0.0f;
    if (col < NCH * EMB) {
        int c = col >> 4, e = col & 15;
        int id = ids[t * NCH + c];
        v = emb[id * EMB + e];
    }
    Xb[idx] = (__bf16)v;
}

// Pack conv weights [oc][EMB][k] -> Bt [oc][K_pad] with Bt[o][dk*16+e] = w[o][e][dk],
// K zero-padded to a multiple of 32.
__global__ void k_prep_convw(const float* __restrict__ w0, const float* __restrict__ w1,
                             const float* __restrict__ w2, const float* __restrict__ w3,
                             const float* __restrict__ w4, const float* __restrict__ w5,
                             const float* __restrict__ w6,
                             __bf16* __restrict__ W) {
    int idx = blockIdx.x * blockDim.x + threadIdx.x;
    if (idx >= CONVW_ELEMS) return;
    const float* w; int base, Kp, ksz;
    if      (idx < 1024)   { w = w0; base = 0;     Kp = 32;  ksz = 1; }
    else if (idx < 2048)   { w = w1; base = 1024;  Kp = 32;  ksz = 2; }
    else if (idx < 6144)   { w = w2; base = 2048;  Kp = 64;  ksz = 3; }
    else if (idx < 14336)  { w = w3; base = 6144;  Kp = 64;  ksz = 4; }
    else if (idx < 38912)  { w = w4; base = 14336; Kp = 96;  ksz = 5; }
    else if (idx < 88064)  { w = w5; base = 38912; Kp = 96;  ksz = 6; }
    else                   { w = w6; base = 88064; Kp = 128; ksz = 7; }
    int local = idx - base;
    int o = local / Kp, j = local - o * Kp;
    float v = 0.0f;
    if (j < EMB * ksz) {
        int dk = j >> 4, e = j & 15;
        v = w[(o * EMB + e) * ksz + dk];
    }
    W[idx] = (__bf16)v;
}

// Plain f32 -> bf16 cast (highway / projection weights; already [N][K] row-major).
__global__ void k_cvt_bf16(const float* __restrict__ s, __bf16* __restrict__ d, int n) {
    int i = blockIdx.x * blockDim.x + threadIdx.x;
    if (i < n) d[i] = (__bf16)s[i];
}

// ---------------- Conv: implicit GEMM + max-over-time + ReLU ----------------
// Template-specialized per kernel size: NK and P are compile-time, so the B
// fragment array is statically indexed (registers, no v_movrels) and the
// position loop unrolls x2 into independent accumulate chains.
// One wave per (16-token tile x 16-filter tile).
template<int KSZ, int KP, int WBASE, int GBASE, int OCT>
__global__ void __launch_bounds__(256)
k_conv_t(const __bf16* __restrict__ Xb, const __bf16* __restrict__ W,
         const float* __restrict__ bias, __bf16* __restrict__ H) {
    constexpr int NK = KP / 32;       // K-steps of 32
    constexpr int P  = 51 - KSZ;      // output positions (VALID)
    int wave = (blockIdx.x * blockDim.x + threadIdx.x) >> 5;
    int lane = threadIdx.x & 31;
    int ttile = wave / OCT;
    int ftile = wave - ttile * OCT;
    if (ttile >= TOKENS / 16) return;

    int half = lane >> 4;
    int nn   = lane & 15;
    int t0   = ttile << 4;
    int n0   = (ftile << 4) + nn;     // filter row within this option's Bt
    const __bf16* arow = Xb + (size_t)(t0 + nn) * XCOL;
    const __bf16* bt   = W + WBASE;
    float bb = bias[n0];

    // Position-invariant B fragments, statically indexed.
    v16bf bfr[NK];
    #pragma unroll
    for (int q = 0; q < NK; ++q)
        bfr[q] = load_b_frag(bt, KP, n0, q << 5, half);

    v8f m;
    #pragma unroll
    for (int j = 0; j < 8; ++j) m[j] = -3.402823466e38f;

    #pragma unroll 2
    for (int p = 0; p < P; ++p) {
        v8f acc = {0.f,0.f,0.f,0.f,0.f,0.f,0.f,0.f};
        #pragma unroll
        for (int q = 0; q < NK; ++q)
            acc = wmma_bf16(load_a_frag(arow, (p << 4) + (q << 5), half),
                            bfr[q], acc);
        #pragma unroll
        for (int j = 0; j < 8; ++j) m[j] = fmaxf(m[j], acc[j]);
    }

    int col = GBASE + (ftile << 4) + nn;
    #pragma unroll
    for (int j = 0; j < 8; ++j) {
        float h = fmaxf(m[j] + bb, 0.0f);                  // relu(max + bias)
        H[(size_t)(t0 + j + half * 8) * NFILT + col] = (__bf16)h;
    }
}

// ---------------- WMMA GEMM: C[M,N] = A[M,K] * Bt[N,K]^T + bias -------------
// Per-wave tile: 32M x 64N -> 8 independent accumulator chains; each B
// fragment feeds 2 WMMAs, A fragments reused across 4 N-tiles. N, K are
// compile-time so the k-loop unrolls (x2) with static fragment indexing.
template<int N, int K>
__global__ void __launch_bounds__(256)
k_gemm_t(const __bf16* __restrict__ A, const __bf16* __restrict__ Bt,
         const float* __restrict__ bias, float* __restrict__ C, int M) {
    constexpr int NT = N >> 6;        // 64-wide N tiles
    int wave = (blockIdx.x * blockDim.x + threadIdx.x) >> 5;
    int lane = threadIdx.x & 31;
    int m0 = (wave / NT) << 5;        // 32 rows per wave
    int n0 = (wave % NT) << 6;        // 64 cols per wave
    if (m0 >= M) return;

    int half = lane >> 4;
    int nn   = lane & 15;
    const __bf16* arow0 = A + (size_t)(m0 + nn) * K;
    const __bf16* arow1 = A + (size_t)(m0 + 16 + nn) * K;

    v8f acc[2][4];
    #pragma unroll
    for (int r = 0; r < 2; ++r)
        #pragma unroll
        for (int s = 0; s < 4; ++s)
            acc[r][s] = (v8f){0.f,0.f,0.f,0.f,0.f,0.f,0.f,0.f};

    #pragma unroll 2
    for (int k0 = 0; k0 < K; k0 += 32) {
        v16bf a0 = load_a_frag(arow0, k0, half);
        v16bf a1 = load_a_frag(arow1, k0, half);
        #pragma unroll
        for (int s = 0; s < 4; ++s) {
            v16bf b = load_b_frag(Bt, K, n0 + (s << 4) + nn, k0, half);
            acc[0][s] = wmma_bf16(a0, b, acc[0][s]);
            acc[1][s] = wmma_bf16(a1, b, acc[1][s]);
        }
    }

    #pragma unroll
    for (int s = 0; s < 4; ++s) {
        float bb = bias[n0 + (s << 4) + nn];
        #pragma unroll
        for (int r = 0; r < 2; ++r) {
            #pragma unroll
            for (int j = 0; j < 8; ++j) {
                size_t row = (size_t)(m0 + r * 16 + j + half * 8) * N;
                C[row + n0 + (s << 4) + nn] = acc[r][s][j] + bb;
            }
        }
    }
}

// ---------------- Highway combine: h = g*h + (1-g)*relu(nl), in-place H -----
__global__ void k_highway(const float* __restrict__ Pb, __bf16* __restrict__ H) {
    int idx = blockIdx.x * blockDim.x + threadIdx.x;
    if (idx >= TOKENS * NFILT) return;
    int t = idx >> 11, k = idx & (NFILT - 1);
    float nl   = Pb[(size_t)t * HWN + k];                // biased in GEMM
    float gpre = Pb[(size_t)t * HWN + NFILT + k];
    float gate = 1.0f / (1.0f + __expf(-gpre));
    float h    = (float)H[idx];
    float o    = gate * h + (1.0f - gate) * fmaxf(nl, 0.0f);
    H[idx] = (__bf16)o;
}

// ---------------------------------------------------------------------------

extern "C" void kernel_launch(void* const* d_in, const int* in_sizes, int n_in,
                              void* d_out, int out_size, void* d_ws, size_t ws_size,
                              hipStream_t stream) {
    (void)in_sizes; (void)n_in; (void)out_size; (void)ws_size;

    const int*   ids = (const int*)d_in[0];
    const float* emb = (const float*)d_in[1];
    const float* cw[7]; const float* cb[7];
    for (int i = 0; i < 7; ++i) {
        cw[i] = (const float*)d_in[2 + 2 * i];
        cb[i] = (const float*)d_in[3 + 2 * i];
    }
    const float* hw_w0  = (const float*)d_in[16];
    const float* hw_b0  = (const float*)d_in[17];
    const float* hw_w1  = (const float*)d_in[18];
    const float* hw_b1  = (const float*)d_in[19];
    const float* proj_w = (const float*)d_in[20];
    const float* proj_b = (const float*)d_in[21];
    float* out = (float*)d_out;

    // Workspace layout (~63 MB total, all fully written each call):
    char* ws = (char*)d_ws;
    __bf16* Xb = (__bf16*)(ws);                       //  3,407,872 B  [2048 x 832]
    __bf16* Wc = (__bf16*)(ws + 3407872);             //    438,272 B  packed conv weights
    __bf16* H  = (__bf16*)(ws + 3846144);             //  8,388,608 B  [2048 x 2048]
    __bf16* Wb = (__bf16*)(ws + 12234752);            // 16,777,216 B  bf16 weight staging
    float*  Pb = (float*) (ws + 29011968);            // 33,554,432 B  [2048 x 4096] f32

    dim3 blk(256);

    // 1. Embedding lookup + layout (bf16)
    k_prep_x<<<(TOKENS * XCOL + 255) / 256, blk, 0, stream>>>(ids, emb, Xb);
    // 2. Pack conv weights
    k_prep_convw<<<(CONVW_ELEMS + 255) / 256, blk, 0, stream>>>(
        cw[0], cw[1], cw[2], cw[3], cw[4], cw[5], cw[6], Wc);

    // 3. Conv (implicit GEMM) + max-over-time + ReLU -> H.
    //    One specialized kernel per kernel size; blocks = 128 * (oc/16) / 8.
    //    template args: KSZ, KP, WBASE(elem), GBASE(filter), OCT(oc/16)
    k_conv_t<1,  32, 0,     0,    2 ><<<  32, blk, 0, stream>>>(Xb, Wc, cb[0], H);
    k_conv_t<2,  32, 1024,  32,   2 ><<<  32, blk, 0, stream>>>(Xb, Wc, cb[1], H);
    k_conv_t<3,  64, 2048,  64,   4 ><<<  64, blk, 0, stream>>>(Xb, Wc, cb[2], H);
    k_conv_t<4,  64, 6144,  128,  8 ><<< 128, blk, 0, stream>>>(Xb, Wc, cb[3], H);
    k_conv_t<5,  96, 14336, 256,  16><<< 256, blk, 0, stream>>>(Xb, Wc, cb[4], H);
    k_conv_t<6,  96, 38912, 512,  32><<< 512, blk, 0, stream>>>(Xb, Wc, cb[5], H);
    k_conv_t<7, 128, 88064, 1024, 64><<<1024, blk, 0, stream>>>(Xb, Wc, cb[6], H);

    // 4. Highway layer 0: waves = (2048/32)*(4096/64) = 4096 -> 512 blocks
    k_cvt_bf16<<<(HWN * NFILT + 255) / 256, blk, 0, stream>>>(hw_w0, Wb, HWN * NFILT);
    k_gemm_t<HWN, NFILT><<<512, blk, 0, stream>>>(H, Wb, hw_b0, Pb, TOKENS);
    k_highway<<<(TOKENS * NFILT + 255) / 256, blk, 0, stream>>>(Pb, H);

    // 5. Highway layer 1 (reuse staging + P buffers)
    k_cvt_bf16<<<(HWN * NFILT + 255) / 256, blk, 0, stream>>>(hw_w1, Wb, HWN * NFILT);
    k_gemm_t<HWN, NFILT><<<512, blk, 0, stream>>>(H, Wb, hw_b1, Pb, TOKENS);
    k_highway<<<(TOKENS * NFILT + 255) / 256, blk, 0, stream>>>(Pb, H);

    // 6. Projection straight into d_out (f32): waves = 64*8 = 512 -> 64 blocks
    k_cvt_bf16<<<(ODIM * NFILT + 255) / 256, blk, 0, stream>>>(proj_w, Wb, ODIM * NFILT);
    k_gemm_t<ODIM, NFILT><<<64, blk, 0, stream>>>(H, Wb, proj_b, out, TOKENS);
}
